// AttentionPooling_46815143526541
// MI455X (gfx1250) — matrix operand order repaired and checked by reading, not compiled, not run on hardware
//
#include <hip/hip_runtime.h>
#include <hip/hip_bf16.h>

#define DDIM 256
#define HDIM 128
#define NSEG 1024

typedef __attribute__((ext_vector_type(16))) __bf16 v16bf;
typedef __attribute__((ext_vector_type(8)))  float  v8f;

union U16bf { unsigned int u[8]; v16bf v; };

__device__ __forceinline__ unsigned short f2bf(float f) {
    unsigned int u = __float_as_uint(f);
    unsigned int r = ((u >> 16) & 1u) + 0x7FFFu;   // round-to-nearest-even
    return (unsigned short)((u + r) >> 16);
}
__device__ __forceinline__ unsigned int fkey(float f) {   // order-preserving float->uint
    unsigned int u = __float_as_uint(f);
    return (u & 0x80000000u) ? ~u : (u | 0x80000000u);
}
__device__ __forceinline__ float fdec(unsigned int k) {
    unsigned int u = (k & 0x80000000u) ? (k & 0x7FFFFFFFu) : ~k;
    return __uint_as_float(u);
}

// ---------------- init per-segment reductions ----------------
__global__ void k_init(unsigned int* __restrict__ segmax, float* __restrict__ segsum) {
    int i = blockIdx.x * blockDim.x + threadIdx.x;
    if (i < NSEG) { segmax[i] = 0u; segsum[i] = 0.0f; }
}

// ---------------- segment boundaries (batch is sorted) ----------------
__global__ void k_bounds(const int* __restrict__ batch, int* __restrict__ segstart, int N) {
    int i = blockIdx.x * blockDim.x + threadIdx.x;
    if (i >= N) return;
    int b = batch[i];
    int prev = (i == 0) ? -1 : batch[i - 1];
    for (int s = prev + 1; s <= b; ++s) segstart[s] = i;
    if (i == N - 1)
        for (int s = b + 1; s <= NSEG; ++s) segstart[s] = N;
}

// ---------------- W1 (fp32 HxD) -> bf16 packed in WMMA B-matrix lane layout ----------
// B 32x16 bf16 layout: lane L holds N=L%16; VGPR v holds K pair (table order, +8 for lanes 16-31).
// Flat index: (((nt*8 + ks)*32 + lane)*8 + v)
__global__ void k_w1(const float* __restrict__ W1, unsigned int* __restrict__ w1bf) {
    int i = blockIdx.x * blockDim.x + threadIdx.x;          // 0 .. 16383
    if (i >= (HDIM / 16) * (DDIM / 32) * 32 * 8) return;
    int v    =  i        & 7;
    int lane = (i >> 3)  & 31;
    int ks   = (i >> 8)  & 7;
    int nt   =  i >> 11;
    int k  = ks * 32 + ((v < 4) ? (2 * v) : (16 + 2 * (v - 4))) + 8 * (lane >> 4); // K index (= W1 column)
    int h0 = nt * 16 + (lane & 15);                          // N index (= W1 row)
    unsigned short lo = f2bf(W1[h0 * DDIM + k]);
    unsigned short hi = f2bf(W1[h0 * DDIM + k + 1]);
    w1bf[i] = (unsigned int)lo | ((unsigned int)hi << 16);
}

// ---------------- pass 1: alpha = tanh(x @ W1^T) @ W2^T + per-seg max --------------
// 128 threads = 4 waves, 16 rows/wave, 64 rows/block. WMMA bf16 16x16x32, f32 acc.
__global__ void __launch_bounds__(128)
k_alpha(const float* __restrict__ x, const int* __restrict__ batch,
        const unsigned int* __restrict__ w1bf, const float* __restrict__ W2,
        float* __restrict__ alpha, unsigned int* __restrict__ segmax, int N) {
    extern __shared__ unsigned short xt[];                   // [64][282] bf16, stride 282
    const int tid = threadIdx.x;
    const int tileBase = blockIdx.x * 64;

    // stage 64x256 fp32 -> bf16 LDS tile, coalesced (row j/2, col (j&1)*128 + tid)
    for (int j = 0; j < 128; ++j) {
        int idx = j * 128 + tid;
        int row = idx >> 8, col = idx & 255;
        int g = tileBase + row;
        float f = (g < N) ? x[(size_t)g * DDIM + col] : 0.0f;
        xt[row * 282 + col] = f2bf(f);
    }
    __syncthreads();

    const int lane = tid & 31;
    const int wave = tid >> 5;
    const int m    = lane & 15;          // row-in-tile (A) / N-in-tile (B,C)
    const int half = lane >> 4;
    const int wr   = wave * 16;          // this wave's local row base

    v8f acc[8] = {};                     // 8 N-tiles of 16x16 f32 accumulators

    #pragma unroll
    for (int ks = 0; ks < 8; ++ks) {
        // A 16x32 bf16 layout (ISA 7.12.2): lane m row; VGPR v -> K pair
        U16bf A;
        #pragma unroll
        for (int v = 0; v < 8; ++v) {
            int kl = ((v < 4) ? (2 * v) : (16 + 2 * (v - 4))) + 8 * half;
            A.u[v] = *(const unsigned int*)&xt[(wr + m) * 282 + ks * 32 + kl];
        }
        #pragma unroll
        for (int nt = 0; nt < 8; ++nt) {
            U16bf B;
            const uint4* bp = (const uint4*)(w1bf + (((nt * 8 + ks) * 32 + lane) * 8));
            uint4 b0 = bp[0], b1 = bp[1];
            B.u[0] = b0.x; B.u[1] = b0.y; B.u[2] = b0.z; B.u[3] = b0.w;
            B.u[4] = b1.x; B.u[5] = b1.y; B.u[6] = b1.z; B.u[7] = b1.w;
            acc[nt] = __builtin_amdgcn_wmma_f32_16x16x32_bf16(
                false, A.v, false, B.v, (short)0, acc[nt], false, false);
        }
    }

    // alpha[row] = sum_h tanh(S[row][h]) * W2[h]
    // C/D layout: VGPR r, lane -> M = r + 8*half, N = lane&15 (h = nt*16 + N)
    float p[8];
    #pragma unroll
    for (int r = 0; r < 8; ++r) p[r] = 0.0f;
    #pragma unroll
    for (int nt = 0; nt < 8; ++nt) {
        float wv = W2[nt * 16 + m];
        #pragma unroll
        for (int r = 0; r < 8; ++r) p[r] += tanhf(acc[nt][r]) * wv;
    }
    #pragma unroll
    for (int off = 1; off < 16; off <<= 1) {
        #pragma unroll
        for (int r = 0; r < 8; ++r) p[r] += __shfl_xor(p[r], off, 16);
    }
    if (m == 0) {
        #pragma unroll
        for (int r = 0; r < 8; ++r) {
            int row = tileBase + wr + r + 8 * half;
            if (row < N) {
                alpha[row] = p[r];
                atomicMax(&segmax[batch[row]], fkey(p[r]));
            }
        }
    }
}

// ---------------- pass 2: e = exp(alpha - segmax); segsum += e ----------------
__global__ void k_expsum(const int* __restrict__ batch, const unsigned int* __restrict__ segmax,
                         float* __restrict__ alpha, float* __restrict__ segsum, int N) {
    int i = blockIdx.x * blockDim.x + threadIdx.x;
    if (i >= N) return;
    int b = batch[i];
    float e = expf(alpha[i] - fdec(segmax[b]));
    alpha[i] = e;                                    // overwrite in place
    atomicAdd(&segsum[b], e);
}

// ---------------- pass 3: z[s,:] = sum_r x[r,:] * e[r]/segsum[s] ----------------
__global__ void __launch_bounds__(256)
k_pool(const float* __restrict__ x, const float* __restrict__ e,
       const float* __restrict__ segsum, const int* __restrict__ segstart,
       float* __restrict__ out) {
    int s = blockIdx.x;
    int t = threadIdx.x;                              // owns output dim t (D==256)
    int start = segstart[s], end = segstart[s + 1];
    float inv = (end > start) ? (1.0f / segsum[s]) : 0.0f;
    float a0 = 0.f, a1 = 0.f, a2 = 0.f, a3 = 0.f;
    int r = start;
    for (; r + 4 <= end; r += 4) {
        float w0 = e[r] * inv, w1 = e[r + 1] * inv, w2 = e[r + 2] * inv, w3 = e[r + 3] * inv;
        a0 += x[(size_t)r * DDIM + t] * w0;
        a1 += x[(size_t)(r + 1) * DDIM + t] * w1;
        a2 += x[(size_t)(r + 2) * DDIM + t] * w2;
        a3 += x[(size_t)(r + 3) * DDIM + t] * w3;
    }
    for (; r < end; ++r) a0 += x[(size_t)r * DDIM + t] * (e[r] * inv);
    out[s * DDIM + t] = (a0 + a1) + (a2 + a3);
}

extern "C" void kernel_launch(void* const* d_in, const int* in_sizes, int n_in,
                              void* d_out, int out_size, void* d_ws, size_t ws_size,
                              hipStream_t stream) {
    const float* x     = (const float*)d_in[0];
    const int*   batch = (const int*)d_in[1];
    const float* W1    = (const float*)d_in[2];
    const float* W2    = (const float*)d_in[3];
    const int N = in_sizes[0] / DDIM;
    float* out = (float*)d_out;

    // workspace carve (base assumed >=256B aligned):
    char* ws = (char*)d_ws;
    unsigned int* w1bf    = (unsigned int*)ws;                         // 16384 u32 (64KB, 16B aligned)
    float*        alpha   = (float*)(ws + 65536);                      // N f32 (reused as e)
    unsigned int* segmax  = (unsigned int*)(ws + 65536 + (size_t)N * 4);
    float*        segsum  = (float*)((char*)segmax + NSEG * 4);
    int*          segstart= (int*)((char*)segsum + NSEG * 4);          // NSEG+1 ints

    k_init  <<<(NSEG + 255) / 256, 256, 0, stream>>>(segmax, segsum);
    k_w1    <<<64, 256, 0, stream>>>(W1, w1bf);
    k_bounds<<<(N + 255) / 256, 256, 0, stream>>>(batch, segstart, N);

    size_t ldsBytes = 64 * 282 * sizeof(unsigned short);               // 36096 B
    k_alpha <<<(N + 63) / 64, 128, ldsBytes, stream>>>(x, batch, w1bf, W2, alpha, segmax, N);

    k_expsum<<<(N + 255) / 256, 256, 0, stream>>>(batch, segmax, alpha, segsum, N);
    k_pool  <<<NSEG, 256, 0, stream>>>(x, alpha, segsum, segstart, out);
}
